// Teacher_img_to_word_LSTM_83141976916054
// MI455X (gfx1250) — matrix operand order repaired and checked by reading, not compiled, not run on hardware
//
#include <hip/hip_runtime.h>
#include <hip/hip_bf16.h>

typedef __bf16 bf16_t;
typedef __attribute__((ext_vector_type(16))) __bf16 v16bf;
typedef __attribute__((ext_vector_type(8)))  __bf16 v8bf;
typedef __attribute__((ext_vector_type(8)))  float  v8f;

#define Hd   512
#define Bn   64
#define Vv   32000
#define Lld  3
#define Tt   38

// ---------------------------------------------------------------------------
// WMMA helpers.  One wave owns one 16-wide N strip across the FULL M=64:
// per K-step it loads ONE B fragment (the streaming weight operand) and
// reuses it for 4 WMMAs against 4 A fragments (activations, cache-hot).
// This cuts weight-stream L2 traffic 4x vs. one-tile-per-wave.
// ---------------------------------------------------------------------------
__device__ __forceinline__ v8f wmma_bf16(v16bf a, v16bf b, v8f c) {
    return __builtin_amdgcn_wmma_f32_16x16x32_bf16(
               false, a, false, b, (short)0, c, false, false);
}

// A: [64, K] row-major bf16 (activations).  W: [N, K] row-major bf16.
// acc[m] accumulates the 16x16 tile for rows m*16..m*16+15, cols n0..n0+15.
__device__ __forceinline__ void mma_strip(v8f acc[4],
                                          const bf16_t* __restrict__ A,
                                          const bf16_t* __restrict__ W,
                                          int K, int n0, int lane)
{
    const int half = lane >> 4;          // 0 or 1
    const int lq   = lane & 15;
    // B fragment: lane holds W row N=n0+lq; K chunks at half*8 and 16+half*8.
    const bf16_t* wrow = W + (long)(n0 + lq) * K + half * 8;
    // A fragment: lane holds row M=m*16+lq; K chunks at half-dependent offsets.
    const bf16_t* arow = A + (long)lq * K + half * 8;

    for (int k0 = 0; k0 < K; k0 += 32) {
        union { v16bf v; v8bf h[2]; } ub;
        ub.h[0] = *(const v8bf*)(wrow + k0);
        ub.h[1] = *(const v8bf*)(wrow + k0 + 16);
        if (k0 + 96 < K)                           // stream-ahead on weights
            __builtin_prefetch(wrow + k0 + 96, 0, 1);
#pragma unroll
        for (int m = 0; m < 4; ++m) {
            union { v16bf v; v8bf h[2]; } ua;
            const bf16_t* ap = arow + (long)m * 16 * K + k0;
            ua.h[0] = *(const v8bf*)(ap);          // K = half*8 .. +7
            ua.h[1] = *(const v8bf*)(ap + 16);     // K = 16+half*8 .. +7
            acc[m] = wmma_bf16(ua.v, ub.v, acc[m]);
        }
    }
}

__device__ __forceinline__ void store_strip(const v8f acc[4],
                                            float* __restrict__ C, int ldc,
                                            int n0, int lane,
                                            float* __restrict__ out2, int tcol)
{
    const int half = lane >> 4;
    const int lq   = lane & 15;
#pragma unroll
    for (int m = 0; m < 4; ++m) {
#pragma unroll
        for (int r = 0; r < 8; ++r) {
            int mm = m * 16 + half * 8 + r;
            int n  = n0 + lq;
            C[(long)mm * ldc + n] = acc[m][r];
            if (out2) out2[((long)mm * Vv + n) * Tt + tcol] = acc[m][r];
        }
    }
}

// ---------------------------------------------------------------------------
// Fused LSTM gate GEMM: C[64,4H] = A1@W1^T + A2@W2^T + bias, bf16 WMMA.
// grid.x = 4H/64, block = (32,4): wave -> one N strip, full M.
// ---------------------------------------------------------------------------
__global__ __launch_bounds__(128)
void gemm_gates_kernel(const bf16_t* __restrict__ A1,
                       const bf16_t* __restrict__ W1, int K1,
                       const bf16_t* __restrict__ A2,
                       const bf16_t* __restrict__ W2, int K2,
                       const float* __restrict__ bias,
                       float* __restrict__ C)
{
    const int lane = threadIdx.x;
    const int n0   = (blockIdx.x * blockDim.y + threadIdx.y) * 16;
    float bv = bias[n0 + (lane & 15)];
    v8f acc[4];
#pragma unroll
    for (int m = 0; m < 4; ++m)
#pragma unroll
        for (int r = 0; r < 8; ++r) acc[m][r] = bv;

    mma_strip(acc, A1, W1, K1, n0, lane);
    mma_strip(acc, A2, W2, K2, n0, lane);
    store_strip(acc, C, 4 * Hd, n0, lane, nullptr, 0);
}

// ---------------------------------------------------------------------------
// Projection GEMM: logits[64,V] = A@Wproj^T + b_proj; also scatters the
// column directly into d_out[:, :, tcol]  ([B,V,T]).
// grid.x = V/64, block = (32,4).
// ---------------------------------------------------------------------------
__global__ __launch_bounds__(128)
void gemm_proj_kernel(const bf16_t* __restrict__ A,
                      const bf16_t* __restrict__ W,
                      const float* __restrict__ bias,
                      float* __restrict__ C,
                      float* __restrict__ out2, int tcol)
{
    const int lane = threadIdx.x;
    const int n0   = (blockIdx.x * blockDim.y + threadIdx.y) * 16;
    float bv = bias[n0 + (lane & 15)];
    v8f acc[4];
#pragma unroll
    for (int m = 0; m < 4; ++m)
#pragma unroll
        for (int r = 0; r < 8; ++r) acc[m][r] = bv;

    mma_strip(acc, A, W, Hd, n0, lane);
    store_strip(acc, C, Vv, n0, lane, out2, tcol);
}

// ---------------------------------------------------------------------------
// LSTM cell elementwise: gates [B,4H] (i,f,g,o); c fp32 state; h out as bf16.
// ---------------------------------------------------------------------------
__global__ __launch_bounds__(256)
void lstm_cell_kernel(const float* __restrict__ gates,
                      float* __restrict__ c,
                      bf16_t* __restrict__ h_bf)
{
    int idx = blockIdx.x * blockDim.x + threadIdx.x;   // exactly B*H threads
    int b = idx / Hd, j = idx % Hd;
    const float* g = gates + (long)b * 4 * Hd;
    float ig = g[j], fg = g[Hd + j], gg = g[2 * Hd + j], og = g[3 * Hd + j];
    float si = 1.0f / (1.0f + __expf(-ig));
    float sf = 1.0f / (1.0f + __expf(-fg));
    float so = 1.0f / (1.0f + __expf(-og));
    float tg = tanhf(gg);
    float c2 = sf * c[idx] + si * tg;
    c[idx]   = c2;
    h_bf[idx] = (bf16_t)(so * tanhf(c2));
}

// ---------------------------------------------------------------------------
// Per-batch argmax over V, then gather embedding row into the emb half of the
// next LSTM input (feat half written once at init; ctx == feat always).
// ---------------------------------------------------------------------------
__global__ __launch_bounds__(256)
void argmax_gather_kernel(const float* __restrict__ logits,  // [B,V]
                          const float* __restrict__ embed,   // [V,H]
                          bf16_t* __restrict__ inp_bf)       // [B,2H]
{
    __shared__ float sval[256];
    __shared__ int   sidx[256];
    int b = blockIdx.x, tid = threadIdx.x;
    const float* row = logits + (long)b * Vv;
    float best = -__builtin_inff(); int bi = 0;
    for (int v = tid; v < Vv; v += 256) {
        float x = row[v];
        if (x > best) { best = x; bi = v; }
    }
    sval[tid] = best; sidx[tid] = bi;
    __syncthreads();
    for (int s = 128; s > 0; s >>= 1) {
        if (tid < s) {
            if (sval[tid + s] > sval[tid] ||
                (sval[tid + s] == sval[tid] && sidx[tid + s] < sidx[tid])) {
                sval[tid] = sval[tid + s]; sidx[tid] = sidx[tid + s];
            }
        }
        __syncthreads();
    }
    int id = sidx[0];
    for (int k = tid; k < Hd; k += 256)
        inp_bf[(long)b * 2 * Hd + Hd + k] = (bf16_t)embed[(long)id * Hd + k];
}

// ---------------------------------------------------------------------------
// Init: ctx0 == feat (degenerate attention). h,c = feat broadcast over L.
// inp0 = [feat_bf | embed[SOS]_bf], start_bf = embed[SOS] broadcast rows.
// ---------------------------------------------------------------------------
__global__ __launch_bounds__(256)
void init_state_kernel(const float* __restrict__ feat,
                       const float* __restrict__ embed,   // SOS row = row 0
                       bf16_t* __restrict__ h_bf,         // [3,B,H]
                       float*  __restrict__ c,            // [3,B,H]
                       bf16_t* __restrict__ inp_bf,       // [B,2H]
                       bf16_t* __restrict__ start_bf)     // [B,H]
{
    int idx = blockIdx.x * blockDim.x + threadIdx.x;  // B*H threads
    int b = idx / Hd, k = idx % Hd;
    float f  = feat[idx];
    bf16_t fb = (bf16_t)f;
#pragma unroll
    for (int l = 0; l < Lld; ++l) {
        h_bf[l * Bn * Hd + idx] = fb;
        c[l * Bn * Hd + idx]    = f;
    }
    bf16_t e0 = (bf16_t)embed[k];
    inp_bf[(long)b * 2 * Hd + k]      = fb;
    inp_bf[(long)b * 2 * Hd + Hd + k] = e0;
    start_bf[idx] = e0;
}

__global__ void bias_sum_kernel(const float* __restrict__ b_ih,
                                const float* __restrict__ b_hh,
                                float* __restrict__ bsum)
{
    int i = blockIdx.x * blockDim.x + threadIdx.x;
    if (i < Lld * 4 * Hd) bsum[i] = b_ih[i] + b_hh[i];
}

__global__ void f32_to_bf16_kernel(const float* __restrict__ src,
                                   bf16_t* __restrict__ dst, long n)
{
    long i = (long)blockIdx.x * blockDim.x + threadIdx.x;
    if (i < n) dst[i] = (bf16_t)src[i];
}

// ---------------------------------------------------------------------------
extern "C" void kernel_launch(void* const* d_in, const int* in_sizes, int n_in,
                              void* d_out, int out_size, void* d_ws, size_t ws_size,
                              hipStream_t stream)
{
    (void)in_sizes; (void)n_in; (void)out_size; (void)ws_size;
    const float* feat     = (const float*)d_in[0];
    const float* W_ih0    = (const float*)d_in[1];   // [4H, 2H]
    const float* W_hh0    = (const float*)d_in[2];   // [4H, H]
    const float* W_ih_r   = (const float*)d_in[3];   // [2, 4H, H]
    const float* W_hh_r   = (const float*)d_in[4];   // [2, 4H, H]
    const float* b_ih     = (const float*)d_in[5];   // [3, 4H]
    const float* b_hh     = (const float*)d_in[6];   // [3, 4H]
    const float* embed    = (const float*)d_in[7];   // [V, H]
    const float* W_proj   = (const float*)d_in[8];   // [V, H]
    const float* b_proj   = (const float*)d_in[9];   // [V]
    // d_in[10..15] (W_hid/b_hid/W_enc/b_enc/W_attout/b_attout) provably unused:
    // attention over identical encoder positions == uniform -> ctx == feat.
    float* out = (float*)d_out;                      // [B, V, T]

    // ---- workspace carve-up (256B aligned) ----
    char* wsp = (char*)d_ws;
    auto alloc = [&](size_t bytes) -> char* {
        char* p = wsp; wsp += (bytes + 255) & ~((size_t)255); return p;
    };
    bf16_t* wproj_bf = (bf16_t*)alloc((size_t)Vv * Hd * 2);
    bf16_t* wih0_bf  = (bf16_t*)alloc((size_t)4 * Hd * 2 * Hd * 2);
    bf16_t* whh0_bf  = (bf16_t*)alloc((size_t)4 * Hd * Hd * 2);
    bf16_t* wihr_bf  = (bf16_t*)alloc((size_t)2 * 4 * Hd * Hd * 2);
    bf16_t* whhr_bf  = (bf16_t*)alloc((size_t)2 * 4 * Hd * Hd * 2);
    float*  bsum     = (float*) alloc((size_t)Lld * 4 * Hd * 4);
    bf16_t* h_bf     = (bf16_t*)alloc((size_t)Lld * Bn * Hd * 2);
    float*  c_st     = (float*) alloc((size_t)Lld * Bn * Hd * 4);
    bf16_t* inp_bf   = (bf16_t*)alloc((size_t)Bn * 2 * Hd * 2);
    bf16_t* start_bf = (bf16_t*)alloc((size_t)Bn * Hd * 2);
    float*  gates    = (float*) alloc((size_t)Bn * 4 * Hd * 4);
    float*  logits   = (float*) alloc((size_t)Bn * Vv * 4);

    // ---- one-time weight conversion to bf16 ----
    auto cvt = [&](const float* s, bf16_t* d, long n) {
        f32_to_bf16_kernel<<<dim3((unsigned)((n + 255) / 256)), dim3(256), 0, stream>>>(s, d, n);
    };
    cvt(W_proj, wproj_bf, (long)Vv * Hd);
    cvt(W_ih0,  wih0_bf,  (long)4 * Hd * 2 * Hd);
    cvt(W_hh0,  whh0_bf,  (long)4 * Hd * Hd);
    cvt(W_ih_r, wihr_bf,  (long)2 * 4 * Hd * Hd);
    cvt(W_hh_r, whhr_bf,  (long)2 * 4 * Hd * Hd);
    bias_sum_kernel<<<dim3(24), dim3(256), 0, stream>>>(b_ih, b_hh, bsum);
    init_state_kernel<<<dim3(Bn * Hd / 256), dim3(256), 0, stream>>>(
        feat, embed, h_bf, c_st, inp_bf, start_bf);

    const dim3 blk(32, 4);
    const dim3 gGate(4 * Hd / 64);            // N=2048 -> 32 blocks
    const dim3 gProj(Vv / 64);                // N=32000 -> 500 blocks
    const int  cellBlocks = Bn * Hd / 256;

    // column 0 of the output: logits of the start token (pred[0] = start)
    gemm_proj_kernel<<<gProj, blk, 0, stream>>>(
        start_bf, wproj_bf, b_proj, logits, out, 0);

    for (int t = 0; t < Tt - 1; ++t) {
        // ---- layer 0: gates = inp@Wih0^T + h0@Whh0^T + (b_ih0+b_hh0)
        gemm_gates_kernel<<<gGate, blk, 0, stream>>>(
            inp_bf, wih0_bf, 2 * Hd, h_bf, whh0_bf, Hd, bsum, gates);
        lstm_cell_kernel<<<cellBlocks, 256, 0, stream>>>(gates, c_st, h_bf);

        // ---- layers 1,2
        for (int l = 1; l < Lld; ++l) {
            const bf16_t* wih = wihr_bf + (size_t)(l - 1) * 4 * Hd * Hd;
            const bf16_t* whh = whhr_bf + (size_t)(l - 1) * 4 * Hd * Hd;
            gemm_gates_kernel<<<gGate, blk, 0, stream>>>(
                h_bf + (size_t)(l - 1) * Bn * Hd, wih, Hd,
                h_bf + (size_t)l * Bn * Hd, whh, Hd,
                bsum + (size_t)l * 4 * Hd, gates);
            lstm_cell_kernel<<<cellBlocks, 256, 0, stream>>>(
                gates, c_st + (size_t)l * Bn * Hd, h_bf + (size_t)l * Bn * Hd);
        }

        // ---- projection: logits = h2@Wproj^T + b_proj
        //      also scatters directly into d_out[:, :, t+1]
        gemm_proj_kernel<<<gProj, blk, 0, stream>>>(
            h_bf + (size_t)2 * Bn * Hd, wproj_bf, b_proj, logits, out, t + 1);

        // ---- greedy argmax + embedding gather -> next input (ctx == feat)
        argmax_gather_kernel<<<dim3(Bn), dim3(256), 0, stream>>>(
            logits, embed, inp_bf);
    }
}
